// VectorQuantizer_51848845197813
// MI455X (gfx1250) — matrix operand order repaired and checked by reading, not compiled, not run on hardware
//
#include <hip/hip_runtime.h>
#include <hip/hip_bf16.h>
#include <math.h>

// ---- shapes ----
#define BATCH   32
#define CH      64          // D (embedding dim)
#define HWSZ    4096        // 64*64
#define NROWS   131072      // BATCH*HWSZ
#define KCB     1024        // codebook size

// d_out element offsets (f32): loss | out(NCHW) | perplexity | encodings | inputs(NHWC)
#define OFF_Q     1
#define OFF_PERP  8388609
#define OFF_ENC   8388610
#define OFF_INP   142606338   // 8388610 + 134217728

typedef __attribute__((ext_vector_type(16))) __bf16 v16bf;
typedef __attribute__((ext_vector_type(8)))  __bf16 v8bf;
typedef __attribute__((ext_vector_type(8)))  float  v8f;

__device__ __forceinline__ __bf16 tobf16(float f) {
    union { float f; unsigned u; } v; v.f = f;
    unsigned r = v.u + 0x7FFFu + ((v.u >> 16) & 1u);   // round-to-nearest-even
    unsigned short h = (unsigned short)(r >> 16);
    return __builtin_bit_cast(__bf16, h);
}

// ---------------- prep: codebook -> bf16, row squared norms ----------------
__global__ __launch_bounds__(256) void vq_prep(const float* __restrict__ cb,
                                               __bf16* __restrict__ cbf,
                                               float* __restrict__ c2) {
    int k = blockIdx.x * 256 + threadIdx.x;           // grid 4 -> k in [0,1024)
    const float* r = cb + (size_t)k * CH;
    float s = 0.f;
#pragma unroll
    for (int d = 0; d < CH; ++d) {
        float v = r[d];
        s += v * v;
        cbf[(size_t)k * CH + d] = tobf16(v);
    }
    c2[k] = s;
}

__global__ __launch_bounds__(256) void vq_zero(int* __restrict__ hist,
                                               float* __restrict__ partial) {
    int t = blockIdx.x * 256 + threadIdx.x;           // grid 4 -> 1024 threads
    if (t < KCB) hist[t] = 0;
    if (t < 512) partial[t] = 0.f;
}

// ---------------- main: WMMA distance GEMM + argmin ----------------
// Block: 256 threads (8 waves). Each block: 128 rows x all 1024 codes.
// Each wave: 16 rows; codebook staged in LDS 256 rows/pass (4 passes).
// B fragments are software-pipelined one 16x16 tile ahead to hide LDS latency
// behind the WMMA issue + argmin fold; ||c||^2 lives in LDS (no VMEM in loop).
__global__ __launch_bounds__(256) void vq_argmin(const float* __restrict__ x,
                                                 const __bf16* __restrict__ cbf,
                                                 const float* __restrict__ c2,
                                                 int* __restrict__ idxg) {
    __shared__ __align__(16) __bf16 sA[128 * 72];     // A tile, padded stride 72 (144B, 16B-aligned)
    __shared__ __align__(16) __bf16 sB[256 * 72];     // codebook chunk
    __shared__ __align__(16) float  sC[KCB];          // ||c||^2

    const int t   = threadIdx.x;
    const int n0  = blockIdx.x * 128;
    const int b   = n0 >> 12;                         // 4096 rows per batch image
    const int hw0 = n0 & 4095;                        // 128 | 4096 -> single image per block

    // Load A tile: x[b][c][hw0+i] -> sA[i*72 + c]  (coalesced along hw)
    const float* xb = x + (size_t)b * CH * HWSZ + hw0;
#pragma unroll
    for (int it = 0; it < 32; ++it) {
        int idx = it * 256 + t;
        int c = idx >> 7;
        int i = idx & 127;
        sA[i * 72 + c] = tobf16(xb[(size_t)c * HWSZ + i]);
    }
    // Stage ||c||^2 into LDS
#pragma unroll
    for (int it = 0; it < 4; ++it) sC[it * 256 + t] = c2[it * 256 + t];
    __syncthreads();

    const int lane = t & 31;
    const int wv   = t >> 5;        // wave 0..7 -> rows [16wv,16wv+16)
    const int lq   = lane & 15;
    const int hi   = lane >> 4;     // half-wave select

    // A fragments, contiguous K order: lanes<16 hold K{0..15}, lanes>=16 K{16..31}
    v16bf a0, a1;
    {
        const __bf16* ar = sA + (wv * 16 + lq) * 72;
        int db = hi ? 16 : 0;
        v8bf p0 = *(const v8bf*)(ar + db);
        v8bf p1 = *(const v8bf*)(ar + db + 8);
        v8bf p2 = *(const v8bf*)(ar + db + 32);
        v8bf p3 = *(const v8bf*)(ar + db + 40);
#pragma unroll
        for (int e = 0; e < 8; ++e) {
            a0[e] = p0[e]; a0[8 + e] = p1[e];
            a1[e] = p2[e]; a1[8 + e] = p3[e];
        }
    }

    // B-fragment loader, same contiguous K order as A: lane = column, contiguous K run
    auto ldB = [&](int kt, v16bf& b0, v16bf& b1) {
        const __bf16* br = sB + (kt * 16 + lq) * 72 + hi * 16;
        v8bf q0 = *(const v8bf*)(br);
        v8bf q1 = *(const v8bf*)(br + 8);
        v8bf q2 = *(const v8bf*)(br + 32);
        v8bf q3 = *(const v8bf*)(br + 40);
#pragma unroll
        for (int e = 0; e < 8; ++e) {
            b0[e] = q0[e]; b0[8 + e] = q1[e];
            b1[e] = q2[e]; b1[8 + e] = q3[e];
        }
    };

    float best[8];
    int   bidx[8];
#pragma unroll
    for (int i = 0; i < 8; ++i) { best[i] = 3.4e38f; bidx[i] = 0; }

    for (int kh = 0; kh < 4; ++kh) {
        __syncthreads();
        // Stage 256 codebook rows (bf16) into LDS; prefetch next chunk.
        const unsigned* src = (const unsigned*)(cbf + (size_t)kh * 256 * CH);
#pragma unroll
        for (int it = 0; it < 32; ++it) {
            int idx = it * 256 + t;                   // dword granularity: 32 dwords per row
            int k  = idx >> 5;
            int dd = (idx & 31) * 2;
            *(unsigned*)(sB + k * 72 + dd) = src[idx];
        }
        if (kh < 3) __builtin_prefetch(cbf + (size_t)(kh + 1) * 256 * CH + t * 64, 0, 3);
        __syncthreads();

        // Software pipeline: B frags + ||c||^2 loaded one tile ahead ((kt+1)&15 wrap keeps loop branch-free)
        v16bf b0c, b1c;
        ldB(0, b0c, b1c);
        float ccur = sC[kh * 256 + lq];

        for (int kt = 0; kt < 16; ++kt) {
            int ktn = (kt + 1) & 15;
            v16bf b0n, b1n;
            ldB(ktn, b0n, b1n);
            float cnext = sC[kh * 256 + ktn * 16 + lq];

            v8f acc = {0.f, 0.f, 0.f, 0.f, 0.f, 0.f, 0.f, 0.f};
            acc = __builtin_amdgcn_wmma_f32_16x16x32_bf16(false, a0, false, b0c,
                                                          (short)0, acc, false, false);
            acc = __builtin_amdgcn_wmma_f32_16x16x32_bf16(false, a1, false, b1c,
                                                          (short)0, acc, false, false);

            int col = kh * 256 + kt * 16 + lq;
#pragma unroll
            for (int i = 0; i < 8; ++i) {
                float d = ccur - 2.0f * acc[i];       // + ||x||^2 is row-constant: dropped
                if (d < best[i]) { best[i] = d; bidx[i] = col; }
            }
            b0c = b0n; b1c = b1n; ccur = cnext;
        }
    }

    // Cross-lane argmin within each 16-lane group (C layout: lanes0-15 -> M=i, lanes16-31 -> M=8+i)
#pragma unroll
    for (int s = 1; s < 16; s <<= 1) {
#pragma unroll
        for (int i = 0; i < 8; ++i) {
            float ov = __shfl_xor(best[i], s, 32);
            int   oi = __shfl_xor(bidx[i], s, 32);
            if (ov < best[i] || (ov == best[i] && oi < bidx[i])) { best[i] = ov; bidx[i] = oi; }
        }
    }
    if (lq == 0) {
        int nb = n0 + wv * 16 + hi * 8;
#pragma unroll
        for (int i = 0; i < 8; ++i) idxg[nb + i] = bidx[i];
    }
}

// ---------------- NCHW -> NHWC "inputs" output ----------------
__global__ __launch_bounds__(256) void vq_transpose(const float* __restrict__ x,
                                                    float* __restrict__ inp) {
    __shared__ float tile[64 * 65];
    int t   = threadIdx.x;
    int b   = blockIdx.x >> 6;                        // grid 2048 = 32 * 64
    int hw0 = (blockIdx.x & 63) * 64;
    const float* xb = x + (size_t)b * CH * HWSZ + hw0;
#pragma unroll
    for (int it = 0; it < 16; ++it) {
        int idx = it * 256 + t;
        int c = idx >> 6, i = idx & 63;
        tile[c * 65 + i] = xb[(size_t)c * HWSZ + i];
    }
    __syncthreads();
    float* ob = inp + ((size_t)b * HWSZ + hw0) * CH;
#pragma unroll
    for (int it = 0; it < 16; ++it) {
        int idx = it * 256 + t;
        int i = idx >> 6, c = idx & 63;
        ob[(size_t)i * CH + c] = tile[c * 65 + i];
    }
}

// ---------------- gather quantized -> out (NCHW), loss partials, histogram ----------------
__global__ __launch_bounds__(256) void vq_gather(const float* __restrict__ x,
                                                 const float* __restrict__ cb,
                                                 const int* __restrict__ idxg,
                                                 float* __restrict__ outq,
                                                 int* __restrict__ hist,
                                                 float* __restrict__ partial) {
    __shared__ float red[256];
    int t  = threadIdx.x;
    int n  = blockIdx.x * 256 + t;
    int b  = n >> 12;
    int hw = n & 4095;
    int my = idxg[n];
    atomicAdd(&hist[my], 1);                          // integer atomics: deterministic
    const float* cbr = cb + (size_t)my * CH;
    size_t xbase = (size_t)b * CH * HWSZ + hw;
    float acc = 0.f;
#pragma unroll
    for (int c = 0; c < CH; ++c) {
        float xv = x[xbase + (size_t)c * HWSZ];
        float qv = cbr[c];
        outq[xbase + (size_t)c * HWSZ] = qv;
        float d = qv - xv;
        acc += d * d;
    }
    red[t] = acc;
    __syncthreads();
#pragma unroll
    for (int s = 128; s > 0; s >>= 1) {
        if (t < s) red[t] += red[t + s];
        __syncthreads();
    }
    if (t == 0) partial[blockIdx.x] = red[0];
}

// ---------------- one-hot encodings (the 537 MB write) ----------------
__global__ __launch_bounds__(256) void vq_onehot(const int* __restrict__ idxg,
                                                 float* __restrict__ enc) {
    int t  = threadIdx.x;
    int wv = t >> 5, j = t & 31;                      // wave per row
    int n  = blockIdx.x * 8 + wv;
    int my = idxg[n];
    float* row = enc + (size_t)n * KCB;               // 8B-aligned base (OFF_ENC is even)
#pragma unroll
    for (int i = 0; i < 16; ++i) {
        int col = i * 64 + j * 2;
        float2 v;
        v.x = (col     == my) ? 1.f : 0.f;
        v.y = (col + 1 == my) ? 1.f : 0.f;
        *(float2*)(row + col) = v;
    }
}

// ---------------- scalars: loss, perplexity ----------------
__global__ void vq_finalize(const int* __restrict__ hist,
                            const float* __restrict__ partial,
                            float* __restrict__ loss_out,
                            float* __restrict__ perp_out) {
    if (threadIdx.x == 0 && blockIdx.x == 0) {
        float s = 0.f;
        for (int i = 0; i < 512; ++i) s += partial[i];         // fixed order: deterministic
        *loss_out = 0.25f * (s / 8388608.0f);                  // mean over N*D
        float e = 0.f;
        for (int k = 0; k < KCB; ++k) {
            float p = (float)hist[k] / (float)NROWS;
            e -= p * logf(p + 1e-10f);
        }
        *perp_out = expf(e);
    }
}

extern "C" void kernel_launch(void* const* d_in, const int* in_sizes, int n_in,
                              void* d_out, int out_size, void* d_ws, size_t ws_size,
                              hipStream_t stream) {
    const float* x  = (const float*)d_in[0];   // [32,64,64,64] NCHW
    const float* cb = (const float*)d_in[1];   // [1024,64]
    float* out = (float*)d_out;

    char* ws = (char*)d_ws;
    __bf16* cbf    = (__bf16*)ws;                                  // 131072 B
    float*  c2     = (float*)(ws + 131072);                        //   4096 B
    int*    idxg   = (int*)  (ws + 131072 + 4096);                 // 524288 B
    int*    hist   = (int*)  (ws + 131072 + 4096 + 524288);        //   4096 B
    float*  parts  = (float*)(ws + 131072 + 4096 + 524288 + 4096); //   2048 B

    float* out_loss = out;
    float* out_q    = out + OFF_Q;
    float* out_perp = out + OFF_PERP;
    float* out_enc  = out + OFF_ENC;
    float* out_inp  = out + OFF_INP;

    vq_zero     <<<4,     256, 0, stream>>>(hist, parts);
    vq_prep     <<<4,     256, 0, stream>>>(cb, cbf, c2);
    vq_argmin   <<<1024,  256, 0, stream>>>(x, cbf, c2, idxg);
    vq_transpose<<<2048,  256, 0, stream>>>(x, out_inp);
    vq_gather   <<<512,   256, 0, stream>>>(x, cb, idxg, out_q, hist, parts);
    vq_onehot   <<<16384, 256, 0, stream>>>(idxg, out_enc);
    vq_finalize <<<1,     64,  0, stream>>>(hist, parts, out_loss, out_perp);
}
